// TransformerBlock_29566554866234
// MI455X (gfx1250) — matrix operand order repaired and checked by reading, not compiled, not run on hardware
//
#include <hip/hip_runtime.h>

// ---------------------------------------------------------------------------
// Transformer block forward for MI455X (gfx1250, wave32).
// GEMMs: TDM (tensor_load_to_lds) stages 128x32 bf16 tiles of A and W into
// double-buffered LDS; waves consume fragments via ds_load_b128 and issue
// v_wmma_f32_16x16x32_bf16. Attention is flash-style, also WMMA.
// ---------------------------------------------------------------------------

typedef __attribute__((ext_vector_type(16))) __bf16       v16bf;
typedef __attribute__((ext_vector_type(8)))  float        v8f;
typedef __attribute__((ext_vector_type(4)))  unsigned int u32x4;
typedef __attribute__((ext_vector_type(8)))  int          i32x8;
typedef __attribute__((ext_vector_type(4)))  int          i32x4;

#define TT  2048   // sequence length T
#define CC  1024   // channels C
#define HH  16     // heads
#define HSZ 64     // head size
#define BB  4      // batch

#if defined(__has_builtin)
#  if __has_builtin(__builtin_amdgcn_tensor_load_to_lds) && \
      __has_builtin(__builtin_amdgcn_s_wait_tensorcnt)
#    define USE_TDM 1
#  endif
#endif
#ifndef USE_TDM
#  define USE_TDM 0
#endif

__device__ __forceinline__ unsigned short f2bf(float f) {
    union { float f; unsigned int u; } v; v.f = f;
    unsigned int r = v.u + 0x7FFFu + ((v.u >> 16) & 1u);   // round-to-nearest-even
    return (unsigned short)(r >> 16);
}

union FragU { v16bf v; u32x4 q[2]; };

// A-matrix fragment, 16x32 bf16, row-major source with leading dim `ld`.
// Lane L<16: row L, K {0..7, 16..23}; lane>=16: row L-16, K {8..15, 24..31}.
__device__ __forceinline__ v16bf load_frag_a(const unsigned short* p, int ld) {
    int l   = threadIdx.x & 31;
    int row = l & 15;
    int k0  = (l < 16) ? 0 : 8;
    const unsigned short* s = p + (size_t)row * ld + k0;
    FragU u;
    u.q[0] = *reinterpret_cast<const u32x4*>(s);        // K k0..k0+7
    u.q[1] = *reinterpret_cast<const u32x4*>(s + 16);   // K k0+16..k0+23
    return u.v;
}

// B-matrix fragment, 32x16 bf16 (KxN). Column n = row n of a row-major [N,K]
// array. Lane<16: K {0..7, 16..23}; lane>=16: K {8..15, 24..31}.
__device__ __forceinline__ v16bf load_frag_b(const unsigned short* p, int ld) {
    int l  = threadIdx.x & 31;
    int n  = l & 15;
    int k0 = (l < 16) ? 0 : 8;
    const unsigned short* s = p + (size_t)n * ld + k0;
    FragU u;
    u.q[0] = *reinterpret_cast<const u32x4*>(s);
    u.q[1] = *reinterpret_cast<const u32x4*>(s + 16);
    return u.v;
}

#define WMMA_BF16(A, B, C) \
    __builtin_amdgcn_wmma_f32_16x16x32_bf16(false, (A), false, (B), (short)0, (C), false, false)

#if USE_TDM
// Issue one TDM 2D tile load: tile_h rows x 32 bf16 elements, row stride
// `strideElems`, into LDS at byte offset `ldsAddr`. D# per ISA 08 section 8.
__device__ __forceinline__ void tdm_load_tile(unsigned ldsAddr,
                                              const unsigned short* g,
                                              int strideElems, int tileH) {
    unsigned long long ga = (unsigned long long)(uintptr_t)g;
    u32x4 g0;
    g0[0] = 1u;                                        // count=1, no gather
    g0[1] = ldsAddr;                                   // lds_addr (bytes)
    g0[2] = (unsigned)(ga & 0xFFFFFFFFull);            // global_addr[31:0]
    g0[3] = ((unsigned)(ga >> 32) & 0x01FFFFFFu) | 0x80000000u;  // [56:32], type=2
    i32x8 g1;
    g1[0] = (1 << 16);                                 // data_size=2B, wg_mask=0
    g1[1] = (int)((unsigned)(strideElems & 0xFFFF) << 16);       // tensor_dim0 lo
    g1[2] = (int)(((unsigned)strideElems >> 16) & 0xFFFFu) |
            (int)((unsigned)(tileH & 0xFFFF) << 16);             // dim0 hi | dim1 lo
    g1[3] = (int)(32u << 16);                          // dim1 hi=0 | tile_dim0=32
    g1[4] = tileH & 0xFFFF;                            // tile_dim1 | tile_dim2=0
    g1[5] = strideElems;                               // tensor_dim0_stride lo
    g1[6] = 0;                                         // stride hi | dim1_stride lo
    g1[7] = 0;
    i32x4 z4 = (i32x4){0, 0, 0, 0};                    // 2D tensor: groups 2/3 off
    i32x8 z8 = (i32x8){0, 0, 0, 0, 0, 0, 0, 0};       // extra group (clang-23 form)
    __builtin_amdgcn_tensor_load_to_lds(g0, g1, z4, z4, z8, 0);
}
#endif

// Cooperative fallback fill: copy tileH x 32 bf16 (row stride strideElems)
// into LDS buffer laid out [tileH][32]. 256 threads.
__device__ __forceinline__ void manual_fill(unsigned short* dst,
                                            const unsigned short* g,
                                            int strideElems, int tileH) {
    int nchunk = tileH * 4;                            // 16B chunks (64B/row)
    for (int c = threadIdx.x; c < nchunk; c += 256) {
        int row = c >> 2, col = (c & 3) * 8;
        *reinterpret_cast<u32x4*>(dst + row * 32 + col) =
            *reinterpret_cast<const u32x4*>(g + (size_t)row * strideElems + col);
    }
}

// ---------------------------------------------------------------------------
__global__ void f2bf_kernel(const float* __restrict__ in,
                            unsigned short* __restrict__ out, int n) {
    int i = blockIdx.x * blockDim.x + threadIdx.x;
    int stride = gridDim.x * blockDim.x;
    for (; i < n; i += stride) out[i] = f2bf(in[i]);
}

// ---------------------------------------------------------------------------
// LayerNorm over C=1024, one 256-thread block per row, bf16 output.
// ---------------------------------------------------------------------------
__global__ __launch_bounds__(256) void ln_bf16_kernel(
        const float* __restrict__ x, const float* __restrict__ g,
        const float* __restrict__ bta, unsigned short* __restrict__ out) {
    int row = blockIdx.x;
    int tid = threadIdx.x;
    float4 v = reinterpret_cast<const float4*>(x + (size_t)row * CC)[tid];
    float s  = v.x + v.y + v.z + v.w;
    float s2 = v.x * v.x + v.y * v.y + v.z * v.z + v.w * v.w;
    for (int off = 1; off < 32; off <<= 1) {
        s  += __shfl_xor(s,  off, 32);
        s2 += __shfl_xor(s2, off, 32);
    }
    __shared__ float red[16];
    int wave = tid >> 5;
    if ((tid & 31) == 0) { red[wave] = s; red[8 + wave] = s2; }
    __syncthreads();
    float S = 0.f, S2 = 0.f;
    for (int i = 0; i < 8; i++) { S += red[i]; S2 += red[8 + i]; }
    float mean = S * (1.0f / CC);
    float var  = S2 * (1.0f / CC) - mean * mean;
    float inv  = rsqrtf(var + 1e-5f);
    int c = tid * 4;
    unsigned short o0 = f2bf((v.x - mean) * inv * g[c + 0] + bta[c + 0]);
    unsigned short o1 = f2bf((v.y - mean) * inv * g[c + 1] + bta[c + 1]);
    unsigned short o2 = f2bf((v.z - mean) * inv * g[c + 2] + bta[c + 2]);
    unsigned short o3 = f2bf((v.w - mean) * inv * g[c + 3] + bta[c + 3]);
    uint2 w;
    w.x = (unsigned)o0 | ((unsigned)o1 << 16);
    w.y = (unsigned)o2 | ((unsigned)o3 << 16);
    *reinterpret_cast<uint2*>(out + (size_t)row * CC + c) = w;
}

// ---------------------------------------------------------------------------
// WMMA GEMM: out = A[M,K] @ W[N,K]^T (+bias)(+epilogue)
// mode 0: bf16 row-major             mode 1: bf16, SiLU(bias added first)
// mode 2: fp32, + bias + residual    mode 3: bf16 transposed V store [B,H,HS,T]
// Block = 256 thr (8 waves, 4x2), tile 128(M)x128(N), K-step 32.
// A and W K-tiles staged through double-buffered LDS; filled by TDM
// (tensor_load_to_lds, overlapped with compute) or cooperative copy fallback.
// ---------------------------------------------------------------------------
__global__ __launch_bounds__(256) void gemm_bf16_kernel(
        const unsigned short* __restrict__ A, const unsigned short* __restrict__ W,
        const float* __restrict__ bias, const float* __restrict__ resid,
        void* __restrict__ outp, int M, int N, int K, int mode) {
    __shared__ __align__(16) unsigned short sA[2][128 * 32];
    __shared__ __align__(16) unsigned short sB[2][128 * 32];

    int wave = threadIdx.x >> 5;
    int lane = threadIdx.x & 31;
    int half = lane >> 4;
    int wr = wave >> 1, wc = wave & 1;
    int mblock = blockIdx.y * 128;
    int nblock = blockIdx.x * 128;
    int mbase = mblock + wr * 32;
    int nbase = nblock + wc * 64;

    const unsigned short* Ab = A + (size_t)mblock * K;
    const unsigned short* Wb = W + (size_t)nblock * K;

#if USE_TDM
    bool issuer = (wave == 0);
    if (issuer) {
        tdm_load_tile((unsigned)(uintptr_t)&sA[0][0], Ab, K, 128);
        tdm_load_tile((unsigned)(uintptr_t)&sB[0][0], Wb, K, 128);
        __builtin_amdgcn_s_wait_tensorcnt(0);
    }
    __syncthreads();
#else
    manual_fill(&sA[0][0], Ab, K, 128);
    manual_fill(&sB[0][0], Wb, K, 128);
    __syncthreads();
#endif

    v8f acc[2][4];
    for (int i = 0; i < 2; i++)
        for (int j = 0; j < 4; j++)
            acc[i][j] = (v8f){0.f, 0.f, 0.f, 0.f, 0.f, 0.f, 0.f, 0.f};

    int nk = K >> 5;
    for (int ki = 0; ki < nk; ki++) {
        int cur = ki & 1;
        // prefetch next K-tile into the other LDS buffer (overlaps WMMAs)
        bool have_next = (ki + 1) < nk;
#if USE_TDM
        if (issuer && have_next) {
            int k1 = (ki + 1) << 5;
            tdm_load_tile((unsigned)(uintptr_t)&sA[cur ^ 1][0], Ab + k1, K, 128);
            tdm_load_tile((unsigned)(uintptr_t)&sB[cur ^ 1][0], Wb + k1, K, 128);
        }
#else
        if (have_next) {
            int k1 = (ki + 1) << 5;
            manual_fill(&sA[cur ^ 1][0], Ab + k1, K, 128);
            manual_fill(&sB[cur ^ 1][0], Wb + k1, K, 128);
        }
#endif
        // compute on current tile (fragments out of LDS)
        const unsigned short* at = &sA[cur][0];
        const unsigned short* bt = &sB[cur][0];
        v16bf a0 = load_frag_a(at + (wr * 32) * 32, 32);
        v16bf a1 = load_frag_a(at + (wr * 32 + 16) * 32, 32);
        for (int j = 0; j < 4; j++) {
            v16bf bf = load_frag_b(bt + (wc * 64 + j * 16) * 32, 32);
            acc[0][j] = WMMA_BF16(a0, bf, acc[0][j]);
            acc[1][j] = WMMA_BF16(a1, bf, acc[1][j]);
        }
#if USE_TDM
        if (issuer && have_next) __builtin_amdgcn_s_wait_tensorcnt(0);
#endif
        __syncthreads();   // next tile published / buffers reusable
    }

    for (int i = 0; i < 2; i++) {
        for (int j = 0; j < 4; j++) {
            for (int r = 0; r < 8; r++) {
                int m = mbase + i * 16 + r + half * 8;
                int n = nbase + j * 16 + (lane & 15);
                float v = acc[i][j][r];
                if (bias) v += bias[n];
                if (mode == 1) v = v / (1.0f + __expf(-v));   // SiLU
                if (mode == 2) {
                    float* o = (float*)outp;
                    size_t idx = (size_t)m * N + n;
                    o[idx] = v + (resid ? resid[idx] : 0.f);
                } else if (mode == 3) {
                    int b_ = m >> 11, t = m & (TT - 1);       // T = 2048
                    int h  = n >> 6,  d = n & (HSZ - 1);      // HS = 64
                    unsigned short* o = (unsigned short*)outp;
                    o[(((size_t)(b_ * HH + h) * HSZ + d) << 11) + t] = f2bf(v);
                } else {
                    unsigned short* o = (unsigned short*)outp;
                    o[(size_t)m * N + n] = f2bf(v);
                }
            }
        }
    }
}

// ---------------------------------------------------------------------------
// Flash attention. Block = 128 thr (4 waves), each wave owns a 16-row q tile.
// q,k: bf16 [B*T, C] (col = h*64+d). v: bf16 transposed [B,H,HS,T].
// Writes xr = x + attn (fp32) — the post-attention residual stream.
// ---------------------------------------------------------------------------
__global__ __launch_bounds__(128) void attn_kernel(
        const unsigned short* __restrict__ qb, const unsigned short* __restrict__ kb,
        const unsigned short* __restrict__ vt, const float* __restrict__ x,
        float* __restrict__ xr) {
    __shared__ __align__(16) unsigned short sP[4][16 * 32];

    const int tilesPerBH = TT / 64;
    int bh   = blockIdx.x / tilesPerBH;
    int tile = blockIdx.x % tilesPerBH;
    int b = bh / HH, h = bh % HH;
    int wave = threadIdx.x >> 5;
    int lane = threadIdx.x & 31;
    int half = lane >> 4;
    int t0 = tile * 64 + wave * 16;

    const unsigned short* qrow = qb + (size_t)(b * TT + t0) * CC + h * HSZ;
    v16bf qa0 = load_frag_a(qrow, CC);        // d = 0..31
    v16bf qa1 = load_frag_a(qrow + 32, CC);   // d = 32..63

    v8f o[4];
    for (int j = 0; j < 4; j++) o[j] = (v8f){0.f,0.f,0.f,0.f,0.f,0.f,0.f,0.f};
    float mst[8], lst[8];
    for (int r = 0; r < 8; r++) { mst[r] = -INFINITY; lst[r] = 0.f; }

    const float scale = 0.125f;  // 1/sqrt(64)
    unsigned short* sp = sP[wave];

    for (int s0 = 0; s0 < t0 + 16; s0 += 32) {
        v8f sa0 = (v8f){0.f,0.f,0.f,0.f,0.f,0.f,0.f,0.f};
        v8f sa1 = sa0;
        const unsigned short* kbase = kb + (size_t)(b * TT + s0) * CC + h * HSZ;
        v16bf k00 = load_frag_b(kbase, CC);
        v16bf k10 = load_frag_b(kbase + (size_t)16 * CC, CC);
        sa0 = WMMA_BF16(qa0, k00, sa0);
        sa1 = WMMA_BF16(qa0, k10, sa1);
        v16bf k01 = load_frag_b(kbase + 32, CC);
        v16bf k11 = load_frag_b(kbase + (size_t)16 * CC + 32, CC);
        sa0 = WMMA_BF16(qa1, k01, sa0);
        sa1 = WMMA_BF16(qa1, k11, sa1);

        float mnew[8];
        int n0 = s0 + (lane & 15);
        for (int r = 0; r < 8; r++) {
            int mrow = t0 + r + half * 8;
            float v0 = (n0      <= mrow) ? sa0[r] * scale : -INFINITY;
            float v1 = (n0 + 16 <= mrow) ? sa1[r] * scale : -INFINITY;
            sa0[r] = v0; sa1[r] = v1;
            mnew[r] = fmaxf(mst[r], fmaxf(v0, v1));
        }
        for (int off = 1; off < 16; off <<= 1)
            for (int r = 0; r < 8; r++)
                mnew[r] = fmaxf(mnew[r], __shfl_xor(mnew[r], off, 32));

        float rs[8];
        for (int r = 0; r < 8; r++) {
            float p0 = __expf(sa0[r] - mnew[r]);
            float p1 = __expf(sa1[r] - mnew[r]);
            sa0[r] = p0; sa1[r] = p1;
            rs[r] = p0 + p1;
        }
        for (int off = 1; off < 16; off <<= 1)
            for (int r = 0; r < 8; r++)
                rs[r] += __shfl_xor(rs[r], off, 32);

        for (int r = 0; r < 8; r++) {
            float alpha = __expf(mst[r] - mnew[r]);
            lst[r] = lst[r] * alpha + rs[r];
            mst[r] = mnew[r];
            for (int j = 0; j < 4; j++) o[j][r] *= alpha;
        }

        // P: D-layout -> A-layout via per-wave LDS tile (16x32 bf16)
        for (int r = 0; r < 8; r++) {
            int m = r + half * 8;
            int c = lane & 15;
            sp[m * 32 + c]      = f2bf(sa0[r]);
            sp[m * 32 + 16 + c] = f2bf(sa1[r]);
        }
        v16bf pa = load_frag_a(sp, 32);   // same-wave LDS ops stay in order

        const unsigned short* vbase = vt + (size_t)(bh * HSZ) * TT + s0;
        for (int j = 0; j < 4; j++) {
            v16bf vb = load_frag_b(vbase + (size_t)(j * 16) * TT, TT);
            o[j] = WMMA_BF16(pa, vb, o[j]);
        }
    }

    for (int j = 0; j < 4; j++) {
        for (int r = 0; r < 8; r++) {
            int t = t0 + r + half * 8;
            int c = h * HSZ + j * 16 + (lane & 15);
            size_t idx = (size_t)(b * TT + t) * CC + c;
            xr[idx] = x[idx] + o[j][r] / lst[r];
        }
    }
}

// ---------------------------------------------------------------------------
extern "C" void kernel_launch(void* const* d_in, const int* in_sizes, int n_in,
                              void* d_out, int out_size, void* d_ws, size_t ws_size,
                              hipStream_t stream) {
    const float* x     = (const float*)d_in[0];
    /* d_in[1] = mask (causal handled analytically) */
    const float* wq    = (const float*)d_in[2];
    const float* wk    = (const float*)d_in[3];
    const float* wv    = (const float*)d_in[4];
    const float* ln1_g = (const float*)d_in[5];
    const float* ln1_b = (const float*)d_in[6];
    const float* ln2_g = (const float*)d_in[7];
    const float* ln2_b = (const float*)d_in[8];
    const float* w1    = (const float*)d_in[9];
    const float* b1    = (const float*)d_in[10];
    const float* w2    = (const float*)d_in[11];
    const float* b2    = (const float*)d_in[12];

    const int BT = BB * TT;            // 8192 token rows
    char* p = (char*)d_ws;
    auto alloc = [&](size_t bytes) -> char* {
        char* r = p; p += (bytes + 255) & ~(size_t)255; return r;
    };
    unsigned short* h1  = (unsigned short*)alloc((size_t)BT * CC * 2);
    unsigned short* wqb = (unsigned short*)alloc((size_t)CC * CC * 2);
    unsigned short* wkb = (unsigned short*)alloc((size_t)CC * CC * 2);
    unsigned short* wvb = (unsigned short*)alloc((size_t)CC * CC * 2);
    unsigned short* w1b = (unsigned short*)alloc((size_t)4 * CC * CC * 2);
    unsigned short* w2b = (unsigned short*)alloc((size_t)4 * CC * CC * 2);
    unsigned short* qb  = (unsigned short*)alloc((size_t)BT * CC * 2);
    unsigned short* kbf = (unsigned short*)alloc((size_t)BT * CC * 2);
    unsigned short* vtb = (unsigned short*)alloc((size_t)BT * CC * 2);   // [B,H,HS,T]
    float*          xrs = (float*)alloc((size_t)BT * CC * 4);
    unsigned short* h2  = (unsigned short*)alloc((size_t)BT * CC * 2);
    unsigned short* ff1 = (unsigned short*)alloc((size_t)BT * 4 * CC * 2);

    f2bf_kernel<<<1024, 256, 0, stream>>>(wq, wqb, CC * CC);
    f2bf_kernel<<<1024, 256, 0, stream>>>(wk, wkb, CC * CC);
    f2bf_kernel<<<1024, 256, 0, stream>>>(wv, wvb, CC * CC);
    f2bf_kernel<<<4096, 256, 0, stream>>>(w1, w1b, 4 * CC * CC);
    f2bf_kernel<<<4096, 256, 0, stream>>>(w2, w2b, 4 * CC * CC);

    ln_bf16_kernel<<<BT, 256, 0, stream>>>(x, ln1_g, ln1_b, h1);

    dim3 blk(256);
    dim3 gqkv(CC / 128, BT / 128);
    gemm_bf16_kernel<<<gqkv, blk, 0, stream>>>(h1, wqb, nullptr, nullptr, qb,  BT, CC, CC, 0);
    gemm_bf16_kernel<<<gqkv, blk, 0, stream>>>(h1, wkb, nullptr, nullptr, kbf, BT, CC, CC, 0);
    gemm_bf16_kernel<<<gqkv, blk, 0, stream>>>(h1, wvb, nullptr, nullptr, vtb, BT, CC, CC, 3);

    attn_kernel<<<BB * HH * (TT / 64), 128, 0, stream>>>(qb, kbf, vtb, x, xrs);

    ln_bf16_kernel<<<BT, 256, 0, stream>>>(xrs, ln2_g, ln2_b, h2);

    dim3 gff1(4 * CC / 128, BT / 128);
    gemm_bf16_kernel<<<gff1, blk, 0, stream>>>(h2, w1b, b1, nullptr, ff1, BT, 4 * CC, CC, 1);
    dim3 gff2(CC / 128, BT / 128);
    gemm_bf16_kernel<<<gff2, blk, 0, stream>>>(ff1, w2b, b2, xrs, d_out, BT, CC, 4 * CC, 2);
}
